// SiVALinear_53523882443032
// MI455X (gfx1250) — matrix operand order repaired and checked by reading, not compiled, not run on hardware
//
#include <hip/hip_runtime.h>
#include <hip/hip_bf16.h>

typedef __bf16 bf16_t;
typedef __attribute__((ext_vector_type(16))) __bf16 v16bf;
typedef __attribute__((ext_vector_type(8)))  __bf16 v8bf;
typedef __attribute__((ext_vector_type(4)))  __bf16 v4bf;
typedef __attribute__((ext_vector_type(8)))  float  v8f;
typedef __attribute__((ext_vector_type(4)))  float  v4f;

typedef __attribute__((ext_vector_type(4))) unsigned int u32x4;
typedef __attribute__((ext_vector_type(8))) int          i32x8;
typedef __attribute__((ext_vector_type(4))) int          i32x4;

static __device__ __forceinline__ v16bf cat8(v8bf lo, v8bf hi) {
  return __builtin_shufflevector(lo, hi, 0,1,2,3,4,5,6,7,8,9,10,11,12,13,14,15);
}

// ---------------------------------------------------------------------------
// Prep: f32 -> bf16 (vectorized x4). Used for x, V_t, V_d.
// ---------------------------------------------------------------------------
__global__ __launch_bounds__(256) void f32_to_bf16_kernel(
    const float* __restrict__ in, bf16_t* __restrict__ out, long n4) {
  long i = (long)blockIdx.x * blockDim.x + threadIdx.x;
  if (i >= n4) return;
  v4f v = *(const v4f*)(in + i * 4);
  v4bf o;
  o.x = (bf16_t)v.x; o.y = (bf16_t)v.y; o.z = (bf16_t)v.z; o.w = (bf16_t)v.w;
  *(v4bf*)(out + i * 4) = o;
}

// ---------------------------------------------------------------------------
// Prep: Ub[o, r] = bf16( U[o, r] * s[r] ), U = concat(U_t [O x 64], U_d [O x 960])
// ---------------------------------------------------------------------------
__global__ __launch_bounds__(256) void build_u_kernel(
    const float* __restrict__ u_t, const float* __restrict__ u_d,
    const float* __restrict__ s, bf16_t* __restrict__ ub, int total4) {
  int i = blockIdx.x * blockDim.x + threadIdx.x;
  if (i >= total4) return;
  int o  = i >> 8;            // row (out-feature)
  int rq = (i & 255) << 2;    // r base (quads never cross the 64 boundary)
  v4f sv = *(const v4f*)(s + rq);
  v4f uv;
  if (rq < 64) uv = *(const v4f*)(u_t + (size_t)o * 64  + rq);
  else         uv = *(const v4f*)(u_d + (size_t)o * 960 + (rq - 64));
  v4bf ov;
  ov.x = (bf16_t)(uv.x * sv.x); ov.y = (bf16_t)(uv.y * sv.y);
  ov.z = (bf16_t)(uv.z * sv.z); ov.w = (bf16_t)(uv.w * sv.w);
  *(v4bf*)(ub + (size_t)i * 4) = ov;
}

// ---------------------------------------------------------------------------
// TDM: issue a 2D tensor_load_to_lds of a 128-row x 64-col bf16 tile.
// D# per CDNA5 ISA ch.8: group0 = {count/flags, lds_addr, global_addr, type},
// group1 = {mask/data_size/pad, tensor dims, tile dims, strides}.
// Pad 4 DWORDs (16B) every 32 DWORDs (128B = one 64-elem row) -> 144B pitch.
// ---------------------------------------------------------------------------
#define TILE_K      64
#define TILE_ROWS   128
#define LDS_PITCH_B 144                       // 128B row + 16B pad
#define TILE_LDS_B  (TILE_ROWS * LDS_PITCH_B) // 18432 bytes

static __device__ __forceinline__ void tdm_load_tile(
    unsigned lds_addr, const bf16_t* gptr, int tensor_k, int tensor_rows) {
  unsigned long long ga = (unsigned long long)(size_t)gptr;
  u32x4 g0;
  g0[0] = 1u;                                     // count=1, not restore, no gather
  g0[1] = lds_addr;                               // LDS byte address
  g0[2] = (unsigned)(ga & 0xFFFFFFFFu);           // global_addr[31:0]
  g0[3] = (unsigned)((ga >> 32) & 0x01FFFFFFu)    // global_addr[56:32]
        | (2u << 30);                             // type = 2 ("image")
  i32x8 g1;
  g1[0] = (int)((1u << 16)                        // data_size = 1 -> 2 bytes
              | (1u << 20)                        // pad_enable
              | (4u << 22)                        // pad_interval: 32 DWORDs (128B)
              | (3u << 25));                      // pad_amount: 4 DWORDs (16B)
  g1[1] = (int)(((unsigned)tensor_k & 0xFFFFu) << 16);             // tensor_dim0 lo
  g1[2] = (int)((((unsigned)tensor_k >> 16) & 0xFFFFu)             // tensor_dim0 hi
              | (((unsigned)tensor_rows & 0xFFFFu) << 16));        // tensor_dim1 lo
  g1[3] = (int)((((unsigned)tensor_rows >> 16) & 0xFFFFu)          // tensor_dim1 hi
              | ((unsigned)TILE_K << 16));                         // tile_dim0 = 64
  g1[4] = (int)((unsigned)TILE_ROWS);                              // tile_dim1 = 128
  g1[5] = (int)tensor_k;                                           // dim0 stride lo
  g1[6] = 0;                                                       // stride hi / dim1 stride lo
  g1[7] = 0;                                                       // dim1 stride hi
  i32x4 z4 = (i32x4){0, 0, 0, 0};                 // groups 2/3 unused (2D tile)
  i32x8 z8 = (i32x8){0, 0, 0, 0, 0, 0, 0, 0};
  __builtin_amdgcn_tensor_load_to_lds(g0, g1, z4, z4, z8, 0);
}

// ---------------------------------------------------------------------------
// GEMM: C[M,N] = A[M,K] * B[N,K]^T (+ bias for f32 output)
// 256 threads = 8 waves; block tile 128x128; wave tile 32x64 (2x4 frags).
// TDM double-buffered LDS staging (TENSORcnt), bf16 WMMA 16x16x32, f32 acc.
// ---------------------------------------------------------------------------
template <bool WRITE_BF16>
__global__ __launch_bounds__(256) void gemm_nt_wmma_kernel(
    const bf16_t* __restrict__ A, const bf16_t* __restrict__ B,
    float* __restrict__ Cf, bf16_t* __restrict__ Cb,
    const float* __restrict__ bias, int M, int N, int K) {
  // 2 buffers x (A tile + B tile), 16B units for conflict-free ds_load_b128
  __shared__ v8bf smem[2 * 2 * TILE_LDS_B / 16];

  const int lane = threadIdx.x & 31;
  const int wave = threadIdx.x >> 5;
  const int wm   = wave & 3;        // wave row     0..3
  const int wn   = wave >> 2;       // wave column  0..1
  const int half = lane >> 4;       // lane half
  const int l16  = lane & 15;

  const int m0 = blockIdx.y * 128 + wm * 32;   // wave M base
  const int n0 = blockIdx.x * 128 + wn * 64;   // wave N base

  const unsigned lds_base = (unsigned)(unsigned long long)(size_t)(void*)smem;

  v8f acc[2][4];
#pragma unroll
  for (int i = 0; i < 2; ++i)
#pragma unroll
    for (int j = 0; j < 4; ++j) acc[i][j] = (v8f){};

  const bf16_t* Ag = A + (size_t)(blockIdx.y * 128) * K;  // block A rows
  const bf16_t* Bg = B + (size_t)(blockIdx.x * 128) * K;  // block B rows

  const int KT = K >> 6;  // number of 64-wide K tiles

  // Prologue: stage tile 0 into buffer 0.
  if (wave == 0) {
    tdm_load_tile(lds_base,              Ag, K, M);
    tdm_load_tile(lds_base + TILE_LDS_B, Bg, K, N);
  }

  for (int kt = 0; kt < KT; ++kt) {
    const int cur = kt & 1;
    if (wave == 0) {
      if (kt + 1 < KT) {
        const int nxt = (kt + 1) & 1;
        const int k0  = (kt + 1) << 6;
        const unsigned nb = lds_base + (unsigned)nxt * (2 * TILE_LDS_B);
        tdm_load_tile(nb,               Ag + k0, K, M);
        tdm_load_tile(nb + TILE_LDS_B,  Bg + k0, K, N);
        __builtin_amdgcn_s_wait_tensorcnt(2);  // tile kt complete, kt+1 in flight
      } else {
        __builtin_amdgcn_s_wait_tensorcnt(0);
      }
    }
    __syncthreads();  // publish staged tile to all waves

    const int a16 = cur * (2 * TILE_LDS_B / 16);           // A tile, 16B units
    const int b16 = a16 + (TILE_LDS_B / 16);               // B tile
#pragma unroll
    for (int c = 0; c < 2; ++c) {        // two 32-wide K cells per staged tile
      v16bf a[2], b[4];
#pragma unroll
      for (int i = 0; i < 2; ++i) {      // A layout: halves at K {0..7,16..23}/{8..15,24..31}
        const int row = wm * 32 + i * 16 + l16;
        const int o   = a16 + row * (LDS_PITCH_B / 16) + c * 4 + half;
        a[i] = cat8(smem[o], smem[o + 2]);
      }
#pragma unroll
      for (int j = 0; j < 4; ++j) {      // B layout: halves at K {0..7,16..23}/{8..15,24..31}
        const int row = wn * 64 + j * 16 + l16;
        const int o   = b16 + row * (LDS_PITCH_B / 16) + c * 4 + half;
        b[j] = cat8(smem[o], smem[o + 2]);
      }
#pragma unroll
      for (int i = 0; i < 2; ++i)
#pragma unroll
        for (int j = 0; j < 4; ++j)
          acc[i][j] = __builtin_amdgcn_wmma_f32_16x16x32_bf16(
              false, a[i], false, b[j], (short)0, acc[i][j], false, false);
    }
    __syncthreads();  // tile consumed; buffer may be overwritten next iter
  }

  // Epilogue. f32 C/D layout: VGPR v -> row M = v + 8*half, col N = l16.
#pragma unroll
  for (int i = 0; i < 2; ++i) {
#pragma unroll
    for (int j = 0; j < 4; ++j) {
      const int n = n0 + j * 16 + l16;
      float bv = 0.f;
      if constexpr (!WRITE_BF16) bv = bias[n];
#pragma unroll
      for (int v = 0; v < 8; ++v) {
        const int m = m0 + i * 16 + half * 8 + v;
        const float val = acc[i][j][v] + bv;
        if constexpr (WRITE_BF16) Cb[(size_t)m * N + n] = (bf16_t)val;
        else                      Cf[(size_t)m * N + n] = val;
      }
    }
  }
}

// ---------------------------------------------------------------------------
// Launch: low-rank path  t = x @ V^T ; out = t @ (U*s)^T + bias
// ---------------------------------------------------------------------------
extern "C" void kernel_launch(void* const* d_in, const int* in_sizes, int n_in,
                              void* d_out, int out_size, void* d_ws, size_t ws_size,
                              hipStream_t stream) {
  const float* x    = (const float*)d_in[0];
  const float* u_t  = (const float*)d_in[1];
  const float* u_d  = (const float*)d_in[2];
  const float* s    = (const float*)d_in[3];
  const float* v_t  = (const float*)d_in[4];
  const float* v_d  = (const float*)d_in[5];
  const float* bias = (const float*)d_in[6];

  const int IN  = 4096, OUT = 4096, R = 1024, RT = 64;
  const int M   = in_sizes[0] / IN;   // B*S = 8192

  // Workspace layout (bf16): xb [M,IN] | Vb [R,IN] | Ub [OUT,R] | t [M,R]
  char* ws = (char*)d_ws;
  bf16_t* xb = (bf16_t*)ws;
  bf16_t* Vb = (bf16_t*)(ws + (size_t)M * IN * 2);
  bf16_t* Ub = (bf16_t*)(ws + (size_t)M * IN * 2 + (size_t)R * IN * 2);
  bf16_t* t  = (bf16_t*)(ws + (size_t)M * IN * 2 + (size_t)R * IN * 2 + (size_t)OUT * R * 2);

  {
    long n4 = (long)M * IN / 4;
    f32_to_bf16_kernel<<<dim3((n4 + 255) / 256), dim3(256), 0, stream>>>(x, xb, n4);
  }
  {
    long n4 = (long)RT * IN / 4;   // V_t -> rows [0,64)
    f32_to_bf16_kernel<<<dim3((n4 + 255) / 256), dim3(256), 0, stream>>>(v_t, Vb, n4);
  }
  {
    long n4 = (long)(R - RT) * IN / 4;   // V_d -> rows [64,1024)
    f32_to_bf16_kernel<<<dim3((n4 + 255) / 256), dim3(256), 0, stream>>>(
        v_d, Vb + (size_t)RT * IN, n4);
  }
  {
    int total4 = OUT * R / 4;   // Ub = (U * s) as bf16
    build_u_kernel<<<dim3((total4 + 255) / 256), dim3(256), 0, stream>>>(
        u_t, u_d, s, Ub, total4);
  }

  // GEMM1: t[M,R] = xb @ Vb^T   (bf16 output, no bias)
  gemm_nt_wmma_kernel<true><<<dim3(R / 128, M / 128), dim3(256), 0, stream>>>(
      xb, Vb, nullptr, t, nullptr, M, R, IN);

  // GEMM2: out[M,OUT] = t @ Ub^T + bias   (f32 output)
  gemm_nt_wmma_kernel<false><<<dim3(OUT / 128, M / 128), dim3(256), 0, stream>>>(
      t, Ub, (float*)d_out, nullptr, bias, M, OUT, R);
}
